// CrossAttentionHead_40261023432751
// MI455X (gfx1250) — matrix operand order repaired and checked by reading, not compiled, not run on hardware
//
#include <hip/hip_runtime.h>
#include <stdint.h>

// ---------------------------------------------------------------------------
// CrossAttentionHead for MI455X (gfx1250, wave32, WMMA bf16 16x16x32)
//   B=8, S=T=2304, C=768, D=512
//   k0: weights f32[512][768] -> bf16 Wt[768][512]
//   k1: Q/K/V projection GEMM. One workgroup covers 16 s-rows x full D=512
//       (8 waves x 4 n-tiles) so x is staged ONCE and xt TWICE. x tiles are
//       staged to LDS with double-buffered GLOBAL_LOAD_ASYNC_TO_LDS
//       (ASYNCcnt + s_wait_asynccnt) when the builtin is available.
//   k2: flash attention; 2304x2304 scores never touch HBM; all-batch bf16
//       K+V = 38 MB sits in the 192 MB L2, so flash K/V re-reads are L2 hits.
//       Online softmax parallelized over all 8 waves (shfl_xor reductions).
// ---------------------------------------------------------------------------

#define SB 2304   // sequence length (48*48)
#define CH 768    // input channels
#define DD 512    // head dim

typedef __bf16 v16bf __attribute__((ext_vector_type(16)));
typedef float  v8f   __attribute__((ext_vector_type(8)));

union Frag16 { v16bf v; uint4 q[2]; uint32_t u[8]; };  // 16 bf16 = 8 VGPRs
union CAcc   { v8f   v; float f[8]; };                 // f32 C/D fragment

__device__ __forceinline__ uint16_t f2bf(float f) {           // RNE f32->bf16
  uint32_t u = __float_as_uint(f);
  u += 0x7FFFu + ((u >> 16) & 1u);
  return (uint16_t)(u >> 16);
}
__device__ __forceinline__ uint32_t pk2bf(float a, float b) { // pack 2 bf16
  uint32_t ua = __float_as_uint(a); ua += 0x7FFFu + ((ua >> 16) & 1u);
  uint32_t ub = __float_as_uint(b); ub += 0x7FFFu + ((ub >> 16) & 1u);
  return (ua >> 16) | (ub & 0xFFFF0000u);
}
__device__ __forceinline__ v8f wmma_bf16(v16bf a, v16bf b, v8f c) {
  return __builtin_amdgcn_wmma_f32_16x16x32_bf16(false, a, false, b,
                                                 (short)0, c, false, false);
}

#if defined(__HIP_DEVICE_COMPILE__) &&                                        \
    __has_builtin(__builtin_amdgcn_global_load_async_to_lds_b32) &&           \
    __has_builtin(__builtin_amdgcn_s_wait_asynccnt)
#define USE_ASYNC_LDS 1
typedef __attribute__((address_space(1))) int as1_int;   // HIP "__device__"
typedef __attribute__((address_space(3))) int as3_int;   // HIP "__shared__"
__device__ __forceinline__ void async_g2l_b32(const float* g, float* l) {
  // AS1 global addr == generic addr; AS3 lds addr == low 32 bits of generic
  __builtin_amdgcn_global_load_async_to_lds_b32(
      (as1_int*)(uintptr_t)g, (as3_int*)(uint32_t)(uintptr_t)l, 0, 0);
}
#else
#define USE_ASYNC_LDS 0
#endif

// --------------------------- kernel 0: weight transpose --------------------
__global__ __launch_bounds__(256) void wtrans_kernel(
    const float* __restrict__ wq, const float* __restrict__ wk,
    const float* __restrict__ wv, uint16_t* __restrict__ wt) {
  int idx = blockIdx.x * 256 + threadIdx.x;       // [3][CH][DD]
  int w   = idx / (CH * DD);
  int rem = idx % (CH * DD);
  int c = rem / DD, d = rem % DD;
  const float* src = (w == 0) ? wq : ((w == 1) ? wk : wv);
  wt[idx] = f2bf(src[(size_t)d * CH + c]);        // wt[w][c][d] = W[d][c]
}

// --------------------------- kernel 1: projections -------------------------
// grid (S/16, B, 3): z = proj ; block 256 (8 waves); wave w -> d in [64w,64w+64)
__global__ __launch_bounds__(256) void proj_kernel(
    const float* __restrict__ x, const float* __restrict__ xt,
    const uint16_t* __restrict__ wt,
    const float* __restrict__ bq, const float* __restrict__ bk,
    const float* __restrict__ bv,
    uint16_t* __restrict__ Qb, uint16_t* __restrict__ Ktb,
    uint16_t* __restrict__ Vb) {
  const int s0   = blockIdx.x * 16;
  const int b    = blockIdx.y;
  const int proj = blockIdx.z;                    // 0=Q 1=K 2=V
  const int tid  = threadIdx.x;
  const int wave = tid >> 5, lane = tid & 31;
  const int m = lane & 15, h = lane >> 4;
  const int dW = wave * 64;                       // this wave's 64 d columns

  const float*    src  = (proj == 0) ? x : xt;
  const uint16_t* w    = wt + (size_t)proj * CH * DD;
  const float*    bias = (proj == 0) ? bq : ((proj == 1) ? bk : bv);

  // per-thread staged elements: e = tid, tid+256 of a 32c x 16s tile
  const int cA = tid >> 4,         sA = tid & 15;
  const int cB = (tid + 256) >> 4, sB = tid & 15;

  CAcc acc[4];
#pragma unroll
  for (int j = 0; j < 4; ++j)
#pragma unroll
    for (int r = 0; r < 8; ++r) acc[j].f[r] = 0.f;

#if USE_ASYNC_LDS
  __shared__ __attribute__((aligned(16))) float    As32[2][512]; // f32 tiles
  __shared__ __attribute__((aligned(16))) uint16_t Asb[2][512];  // bf16 tiles

  // issue tile 0 into buffer 0 (2 async b32 per thread)
  {
    async_g2l_b32(&src[((size_t)b * CH + cA) * SB + s0 + sA], &As32[0][sA * 32 + cA]);
    async_g2l_b32(&src[((size_t)b * CH + cB) * SB + s0 + sB], &As32[0][sB * 32 + cB]);
  }
  for (int kc = 0; kc < CH / 32; ++kc) {
    const int bufi = kc & 1;
    if (kc + 1 < CH / 32) {
      const int c0n = (kc + 1) * 32;
      async_g2l_b32(&src[((size_t)b * CH + c0n + cA) * SB + s0 + sA],
                    &As32[bufi ^ 1][sA * 32 + cA]);
      async_g2l_b32(&src[((size_t)b * CH + c0n + cB) * SB + s0 + sB],
                    &As32[bufi ^ 1][sB * 32 + cB]);
      __builtin_amdgcn_s_wait_asynccnt(2);   // current tile done, next in flight
    } else {
      __builtin_amdgcn_s_wait_asynccnt(0);
    }
    __syncthreads();
    // cooperative f32 -> bf16 conversion (one conversion per element total)
    Asb[bufi][tid]       = f2bf(As32[bufi][tid]);
    Asb[bufi][tid + 256] = f2bf(As32[bufi][tid + 256]);
    __syncthreads();

    Frag16 a;
    const uint16_t* ap = &Asb[bufi][m * 32 + 8 * h];
    a.q[0] = *(const uint4*)ap;
    a.q[1] = *(const uint4*)(ap + 16);
    const int c0 = kc * 32;
#pragma unroll
    for (int j = 0; j < 4; ++j) {
      Frag16 bw;
      const uint16_t* bp = w + (size_t)(c0 + 8 * h) * DD + dW + 16 * j + m;
#pragma unroll
      for (int s = 0; s < 16; ++s)
        ((uint16_t*)bw.u)[s] = bp[(size_t)((s & 7) + 16 * (s >> 3)) * DD];
      acc[j].v = wmma_bf16(a.v, bw.v, acc[j].v);
    }
  }
#else
  __shared__ __attribute__((aligned(16))) uint16_t Asb1[512];
  for (int kc = 0; kc < CH / 32; ++kc) {
    const int c0 = kc * 32;
    __syncthreads();
    Asb1[sA * 32 + cA] = f2bf(src[((size_t)b * CH + c0 + cA) * SB + s0 + sA]);
    Asb1[sB * 32 + cB] = f2bf(src[((size_t)b * CH + c0 + cB) * SB + s0 + sB]);
    __syncthreads();
    Frag16 a;
    const uint16_t* ap = &Asb1[m * 32 + 8 * h];
    a.q[0] = *(const uint4*)ap;
    a.q[1] = *(const uint4*)(ap + 16);
#pragma unroll
    for (int j = 0; j < 4; ++j) {
      Frag16 bw;
      const uint16_t* bp = w + (size_t)(c0 + 8 * h) * DD + dW + 16 * j + m;
#pragma unroll
      for (int s = 0; s < 16; ++s)
        ((uint16_t*)bw.u)[s] = bp[(size_t)((s & 7) + 16 * (s >> 3)) * DD];
      acc[j].v = wmma_bf16(a.v, bw.v, acc[j].v);
    }
  }
#endif

  if (proj == 0) {                                // Q[b][s][d], pre-scaled
    const float SCALE = 0.04419417382415922f;     // 1/sqrt(512)
    uint16_t* q = Qb + ((size_t)b * SB) * DD;
#pragma unroll
    for (int j = 0; j < 4; ++j) {
      const float bv_ = bias[dW + 16 * j + m];
#pragma unroll
      for (int r = 0; r < 8; ++r)
        q[(size_t)(s0 + r + 8 * h) * DD + dW + 16 * j + m] =
            f2bf((acc[j].f[r] + bv_) * SCALE);
    }
  } else if (proj == 1) {                         // Kt[b][d][t]
    uint16_t* kt = Ktb + ((size_t)b * DD) * SB;
#pragma unroll
    for (int j = 0; j < 4; ++j) {
      const float bv_ = bias[dW + 16 * j + m];
      const size_t base = (size_t)(dW + 16 * j + m) * SB + s0 + 8 * h;
#pragma unroll
      for (int r = 0; r < 8; r += 2)
        *(uint32_t*)(kt + base + r) =
            pk2bf(acc[j].f[r] + bv_, acc[j].f[r + 1] + bv_);
    }
  } else {                                        // V[b][t][d]
    uint16_t* v = Vb + ((size_t)b * SB) * DD;
#pragma unroll
    for (int j = 0; j < 4; ++j) {
      const float bv_ = bias[dW + 16 * j + m];
#pragma unroll
      for (int r = 0; r < 8; ++r)
        v[(size_t)(s0 + r + 8 * h) * DD + dW + 16 * j + m] =
            f2bf(acc[j].f[r] + bv_);
    }
  }
}

// --------------------------- kernel 2: flash attention ---------------------
// grid (S/16, B); block 256 = 8 waves; wave w owns d slice [64w, 64w+64)
__global__ __launch_bounds__(256) void attn_kernel(
    const uint16_t* __restrict__ Q, const uint16_t* __restrict__ Kt,
    const uint16_t* __restrict__ V, float* __restrict__ out) {
  const int s0  = blockIdx.x * 16;
  const int b   = blockIdx.y;
  const int tid = threadIdx.x;
  const int wave = tid >> 5, lane = tid & 31;
  const int m = lane & 15, h = lane >> 4;
  const int dW = wave * 64;

  __shared__ __attribute__((aligned(16))) float    Ssc[16 * 64]; // f32 scores
  __shared__ __attribute__((aligned(16))) uint16_t Psc[16 * 64]; // bf16 P
  __shared__ __attribute__((aligned(16))) float mRow[16];
  __shared__ __attribute__((aligned(16))) float lRow[16];
  __shared__ __attribute__((aligned(16))) float aRow[16];

  const uint16_t* Qb = Q  + ((size_t)b * SB) * DD;
  const uint16_t* Kb = Kt + ((size_t)b * DD) * SB;
  const uint16_t* Vb = V  + ((size_t)b * SB) * DD;

  // Q A-fragments for this wave's 64-d slice: 2 x (16x32), kept in VGPRs
  Frag16 qf[2];
#pragma unroll
  for (int kc = 0; kc < 2; ++kc) {
    const uint16_t* qp = Qb + (size_t)(s0 + m) * DD + dW + 32 * kc + 8 * h;
    qf[kc].q[0] = *(const uint4*)qp;
    qf[kc].q[1] = *(const uint4*)(qp + 16);
  }

  CAcc acc[4];
#pragma unroll
  for (int j = 0; j < 4; ++j)
#pragma unroll
    for (int r = 0; r < 8; ++r) acc[j].f[r] = 0.f;

  if (tid < 16) { mRow[tid] = -3.0e38f; lRow[tid] = 0.f; }
  __syncthreads();

  const float LOG2E = 1.4426950408889634f;

  for (int tc = 0; tc < SB / 64; ++tc) {
    const int t0 = tc * 64;
    // zero chunk score buffer (1024 floats / 256 threads)
#pragma unroll
    for (int i = 0; i < 4; ++i) Ssc[tid + 256 * i] = 0.f;
    __syncthreads();

    // partial scores over this wave's d slice; reduce across waves in LDS
#pragma unroll
    for (int j = 0; j < 4; ++j) {
      CAcc sf;
#pragma unroll
      for (int r = 0; r < 8; ++r) sf.f[r] = 0.f;
#pragma unroll
      for (int kc = 0; kc < 2; ++kc) {
        Frag16 kb;
        const uint16_t* kp =
            Kb + (size_t)(dW + 32 * kc + 8 * h) * SB + t0 + 16 * j + m;
#pragma unroll
        for (int s = 0; s < 16; ++s)
          ((uint16_t*)kb.u)[s] = kp[(size_t)((s & 7) + 16 * (s >> 3)) * SB];
        sf.v = wmma_bf16(qf[kc].v, kb.v, sf.v);
      }
#pragma unroll
      for (int r = 0; r < 8; ++r)
        atomicAdd(&Ssc[(r + 8 * h) * 64 + 16 * j + m], sf.f[r]); // ds_add_f32
    }
    __syncthreads();

    // online softmax, parallel over all 8 waves:
    //   wave w handles rows {2w, 2w+1}; 16-lane half per row; 4 cols per lane
    {
      const int row = 2 * wave + h;
      float4 s4 = *(const float4*)&Ssc[row * 64 + 4 * m];
      float mc = fmaxf(fmaxf(s4.x, s4.y), fmaxf(s4.z, s4.w));
#pragma unroll
      for (int off = 1; off < 16; off <<= 1)
        mc = fmaxf(mc, __shfl_xor(mc, off, 16));        // ds-permute reduce
      const float mold = mRow[row];
      const float mnew = fmaxf(mold, mc);
      float p0 = __builtin_amdgcn_exp2f((s4.x - mnew) * LOG2E);
      float p1 = __builtin_amdgcn_exp2f((s4.y - mnew) * LOG2E);
      float p2 = __builtin_amdgcn_exp2f((s4.z - mnew) * LOG2E);
      float p3 = __builtin_amdgcn_exp2f((s4.w - mnew) * LOG2E);
      float sum = (p0 + p1) + (p2 + p3);
#pragma unroll
      for (int off = 1; off < 16; off <<= 1)
        sum += __shfl_xor(sum, off, 16);
      uint32_t* pp = (uint32_t*)&Psc[row * 64 + 4 * m];
      pp[0] = pk2bf(p0, p1);
      pp[1] = pk2bf(p2, p3);
      if (m == 0) {
        const float al = __builtin_amdgcn_exp2f((mold - mnew) * LOG2E);
        mRow[row] = mnew;
        aRow[row] = al;
        lRow[row] = lRow[row] * al + sum;
      }
    }
    __syncthreads();

    // rescale accumulators by alpha[row]
    float4 a0 = *(const float4*)&aRow[8 * h];
    float4 a1 = *(const float4*)&aRow[8 * h + 4];
    float al8[8] = {a0.x, a0.y, a0.z, a0.w, a1.x, a1.y, a1.z, a1.w};
#pragma unroll
    for (int j = 0; j < 4; ++j)
#pragma unroll
      for (int r = 0; r < 8; ++r) acc[j].f[r] *= al8[r];

    // P (LDS) x V (global) into accumulators
    Frag16 pf[2];
#pragma unroll
    for (int kc = 0; kc < 2; ++kc) {
      const uint16_t* pq = Psc + m * 64 + 32 * kc + 8 * h;
      pf[kc].q[0] = *(const uint4*)pq;                 // ds_load_b128
      pf[kc].q[1] = *(const uint4*)(pq + 16);
    }
#pragma unroll
    for (int j = 0; j < 4; ++j) {
#pragma unroll
      for (int kc = 0; kc < 2; ++kc) {
        Frag16 vbf;
        const uint16_t* vp =
            Vb + (size_t)(t0 + 32 * kc + 8 * h) * DD + dW + 16 * j + m;
#pragma unroll
        for (int s = 0; s < 16; ++s)
          ((uint16_t*)vbf.u)[s] = vp[(size_t)((s & 7) + 16 * (s >> 3)) * DD];
        acc[j].v = wmma_bf16(pf[kc].v, vbf.v, acc[j].v);
      }
    }
    // prefetch next chunk's K/V cachelines (global_prefetch_b8)
    if (tc + 1 < SB / 64) {
      __builtin_prefetch(Kb + (size_t)(dW + m + 16 * h) * SB + t0 + 64, 0, 1);
      __builtin_prefetch(Vb + (size_t)(t0 + 64 + m + 16 * h) * DD + dW, 0, 1);
    }
    __syncthreads();   // protects Psc/Ssc reuse across iterations
  }

  // normalize by l and store f32 output
  float4 l0 = *(const float4*)&lRow[8 * h];
  float4 l1 = *(const float4*)&lRow[8 * h + 4];
  float li8[8] = {l0.x, l0.y, l0.z, l0.w, l1.x, l1.y, l1.z, l1.w};
  float inv[8];
#pragma unroll
  for (int r = 0; r < 8; ++r) inv[r] = 1.0f / li8[r];
  float* ob = out + ((size_t)b * SB + s0) * DD + dW;
#pragma unroll
  for (int j = 0; j < 4; ++j)
#pragma unroll
    for (int r = 0; r < 8; ++r)
      ob[(size_t)(r + 8 * h) * DD + 16 * j + m] = acc[j].f[r] * inv[r];
}

// --------------------------- host launcher ---------------------------------
extern "C" void kernel_launch(void* const* d_in, const int* in_sizes, int n_in,
                              void* d_out, int out_size, void* d_ws,
                              size_t ws_size, hipStream_t stream) {
  const float* x  = (const float*)d_in[0];
  const float* xt = (const float*)d_in[1];
  const float* wq = (const float*)d_in[2];
  const float* bq = (const float*)d_in[3];
  const float* wk = (const float*)d_in[4];
  const float* bk = (const float*)d_in[5];
  const float* wv = (const float*)d_in[6];
  const float* bv = (const float*)d_in[7];
  float* out = (float*)d_out;

  // workspace layout (bytes), all 16B-aligned:
  //   wt  bf16 [3][768][512]  @ 0          (2,359,296)
  //   Q   bf16 [8][2304][512] @ 2359296    (18,874,368)
  //   Kt  bf16 [8][512][2304] @ 21233664   (18,874,368)
  //   V   bf16 [8][2304][512] @ 40108032   (18,874,368)  -> end 58,982,400
  char* ws = (char*)d_ws;
  uint16_t* wt  = (uint16_t*)(ws);
  uint16_t* Qb  = (uint16_t*)(ws + 2359296);
  uint16_t* Ktb = (uint16_t*)(ws + 21233664);
  uint16_t* Vb  = (uint16_t*)(ws + 40108032);

  wtrans_kernel<<<dim3((3 * CH * DD) / 256), dim3(256), 0, stream>>>(
      wq, wk, wv, wt);
  proj_kernel<<<dim3(SB / 16, 8, 3), dim3(256), 0, stream>>>(
      x, xt, wt, bq, bk, bv, Qb, Ktb, Vb);
  attn_kernel<<<dim3(SB / 16, 8, 1), dim3(256), 0, stream>>>(
      Qb, Ktb, Vb, out);
}
